// VoxelHashTableFlowTraverse_10033043603486
// MI455X (gfx1250) — matrix operand order repaired and checked by reading, not compiled, not run on hardware
//
#include <hip/hip_runtime.h>
#include <hip/hip_bf16.h>

// ---------------------------------------------------------------------------
// VoxelHashTableFlowTraverse for MI455X (gfx1250, wave32, WMMA bf16)
//   k_pack     : pack Wq/Wk/Wv/Wo (120x120 f32) -> 128x128 bf16 B-operand tiles
//   k_timeproj : TQ/TK/TV[t] = time_embeddings[t] @ W{q,k,v} + b  (201 rows)
//   k_gather   : one wave per point: 8-corner hash gather + trilinear blend
//   k_main     : single-wave workgroups; weights async-copied to LDS once per
//                block (global_load_async_to_lds_b128); per 16-point tile:
//                WMMA X@W{q,k,v} (B from LDS), attention in VALU, WMMA Z@Wo.
// ---------------------------------------------------------------------------

typedef __attribute__((ext_vector_type(16))) __bf16 v16bf;
typedef __attribute__((ext_vector_type(8)))  float  v8f;

#define TABLE_SZ  (1 << 20)
#define MOD_T     201
#define D_IN      120
#define DP        128
#define DH        15
#define NTILE_N   8                      // 128 / 16
#define NTILE_K   4                      // 128 / 32
#define WMAT_ELEMS (NTILE_K * NTILE_N * 32 * 16)   // 16384 bf16 per matrix (32 KB)

__device__ __forceinline__ unsigned short bf_bits(float f) {
    union { __bf16 b; unsigned short s; } v; v.b = (__bf16)f; return v.s;
}

// --------------------------------------------------------------------------
// Pack a 120x120 row-major f32 W into B-operand bf16 tiles, zero padded to
// 128x128.  Elem (((kt*8+nt)*32+lane)*16+j) = W[kt*32+(lane>>4)*16+j][nt*16+(lane&15)]
// --------------------------------------------------------------------------
__global__ void k_pack(const float* __restrict__ Wq, const float* __restrict__ Wk,
                       const float* __restrict__ Wv, const float* __restrict__ Wo,
                       unsigned short* __restrict__ packed) {
    int tid = blockIdx.x * blockDim.x + threadIdx.x;
    if (tid >= 4 * NTILE_K * NTILE_N * 32) return;
    int lane = tid & 31;
    int nt   = (tid >> 5) & 7;
    int kt   = (tid >> 8) & 3;
    int mat  = tid >> 10;
    const float* W = (mat == 0) ? Wq : (mat == 1) ? Wk : (mat == 2) ? Wv : Wo;
    unsigned short* dst = packed + ((size_t)((mat * NTILE_K + kt) * NTILE_N + nt) * 32 + lane) * 16;
    int n     = nt * 16 + (lane & 15);
    int kbase = kt * 32 + (lane >> 4) * 16;
#pragma unroll
    for (int j = 0; j < 16; ++j) {
        int k = kbase + j;
        float v = (k < D_IN && n < D_IN) ? W[k * D_IN + n] : 0.0f;
        dst[j] = bf_bits(v);
    }
}

// --------------------------------------------------------------------------
// TQ/TK/TV[t][d] = sum_k emb[t][k] * W[k][d] + b[d]   (201x120 each, tiny)
// --------------------------------------------------------------------------
__global__ void k_timeproj(const float* __restrict__ emb,
                           const float* __restrict__ Wq, const float* __restrict__ bq,
                           const float* __restrict__ Wk, const float* __restrict__ bk,
                           const float* __restrict__ Wv, const float* __restrict__ bv,
                           float* __restrict__ TQ, float* __restrict__ TK, float* __restrict__ TV) {
    int tid = blockIdx.x * blockDim.x + threadIdx.x;
    if (tid >= 3 * MOD_T * D_IN) return;
    int dim = tid % D_IN;
    int t   = (tid / D_IN) % MOD_T;
    int mat = tid / (D_IN * MOD_T);
    const float* W = (mat == 0) ? Wq : (mat == 1) ? Wk : Wv;
    const float* b = (mat == 0) ? bq : (mat == 1) ? bk : bv;
    const float* e = emb + t * D_IN;
    float s = b[dim];
    for (int k = 0; k < D_IN; ++k) s += e[k] * W[k * D_IN + dim];
    float* T = (mat == 0) ? TQ : (mat == 1) ? TK : TV;
    T[t * D_IN + dim] = s;
}

// --------------------------------------------------------------------------
// One wave per point: 8-corner hash gather, trilinear blend, bf16 X row.
// Lane l (l<30) owns dims 4l..4l+3.  h = dot(corner,primes) & (2^20-1)
// (power-of-two modulus == low bits in two's complement, matches jnp.remainder)
// --------------------------------------------------------------------------
__global__ void k_gather(const float* __restrict__ pts, const int* __restrict__ times,
                         const float* __restrict__ dyn, const int* __restrict__ buf,
                         unsigned short* __restrict__ X, int* __restrict__ tmod, int m) {
    int wave = (blockIdx.x * blockDim.x + threadIdx.x) >> 5;
    int lane = threadIdx.x & 31;
    if (wave >= m) return;

    float px = pts[wave * 3 + 0], py = pts[wave * 3 + 1], pz = pts[wave * 3 + 2];
    float gx = px / 0.1f, gy = py / 0.1f, gz = pz / 0.1f;
    float fx = floorf(gx), fy = floorf(gy), fz = floorf(gz);
    float frx = gx - fx, fry = gy - fy, frz = gz - fz;
    int bx = (int)fx, by = (int)fy, bz = (int)fz;

    float a0 = 0.f, a1 = 0.f, a2 = 0.f, a3 = 0.f;
    int col = lane * 4;

#pragma unroll
    for (int c = 0; c < 8; ++c) {
        int ox = c & 1, oy = (c >> 1) & 1, oz = (c >> 2) & 1;
        long long s = (long long)(bx + ox) * 73856093LL
                    + (long long)(by + oy) * 19349669LL
                    + (long long)(bz + oz) * 83492791LL;
        int h = (int)(s & (long long)(TABLE_SZ - 1));
        int vidx = buf[h];
        float wx = ox ? frx : 1.0f - frx;
        float wy = oy ? fry : 1.0f - fry;
        float wz = oz ? frz : 1.0f - frz;
        float w = wx * wy * wz;
        if (vidx >= 0 && col < D_IN) {
            const float4 f = *(const float4*)(dyn + (size_t)vidx * D_IN + col);
            a0 += w * f.x; a1 += w * f.y; a2 += w * f.z; a3 += w * f.w;
        }
    }
    unsigned short* xr = X + (size_t)wave * DP;
    if (col < D_IN) {
        union { __bf16 b[4]; uint2 u; } pk;
        pk.b[0] = (__bf16)a0; pk.b[1] = (__bf16)a1;
        pk.b[2] = (__bf16)a2; pk.b[3] = (__bf16)a3;
        *(uint2*)(xr + col) = pk.u;
    } else if (lane == 30) {           // zero the 8 pad columns (120..127)
        uint4 z; z.x = z.y = z.z = z.w = 0u;
        *(uint4*)(xr + D_IN) = z;
    }
    if (lane == 0) tmod[wave] = times[wave] % MOD_T;
}

// --------------------------------------------------------------------------
// Main fused kernel: 32 threads = ONE wave per workgroup (no barriers needed;
// s_wait_asynccnt alone orders the async weight copy against DS reads).
// LDS: 128 KB packed weights (async-copied once) + 16 KB bf16 staging.
// --------------------------------------------------------------------------
__global__ __launch_bounds__(32) void k_main(
        const unsigned short* __restrict__ X, const int* __restrict__ tmod,
        const unsigned short* __restrict__ packed,
        const float* __restrict__ TQ, const float* __restrict__ TK, const float* __restrict__ TV,
        const float* __restrict__ bq, const float* __restrict__ bk, const float* __restrict__ bv,
        const float* __restrict__ bo, float* __restrict__ out, int m) {
    __shared__ unsigned short sW[4 * WMAT_ELEMS];    // 128 KB: packed Wq,Wk,Wv,Wo
    __shared__ unsigned short sQ[16 * DP];
    __shared__ unsigned short sK[16 * DP];
    __shared__ unsigned short sV[16 * DP];
    __shared__ unsigned short sZ[16 * DP];

    const int lane = threadIdx.x & 31;
    const int ntiles = (m + 15) >> 4;
    const int klo = (lane >> 4) * 16;       // A-operand chunk base within K-window

    // ---- async copy of all packed weights (131072 B) into LDS ------------
    {
        unsigned long long g = (unsigned long long)(const void*)packed + (unsigned)(lane * 16);
        unsigned           l = (unsigned)(size_t)(void*)&sW[0] + (unsigned)(lane * 16);
        for (int i = 0; i < 256; ++i) {     // 256 x (32 lanes x 16 B) = 128 KB
            asm volatile("global_load_async_to_lds_b128 %0, %1, off"
                         :: "v"(l), "v"(g) : "memory");
            g += 512; l += 512;
        }
        asm volatile("s_wait_asynccnt 0x0" ::: "memory");
    }

    for (int tile = blockIdx.x; tile < ntiles; tile += gridDim.x) {
        const int row0 = tile * 16;

        // ---- load A tiles of X (16 rows x 128 K, bf16) -------------------
        v16bf A[NTILE_K];
        {
            int arow = row0 + (lane & 15);
            if (arow >= m) arow = m - 1;             // tail clamp (stores guarded)
            const unsigned short* xr = X + (size_t)arow * DP;
#pragma unroll
            for (int kt = 0; kt < NTILE_K; ++kt) {
                union { uint4 u[2]; v16bf v; } t;
                t.u[0] = *(const uint4*)(xr + kt * 32 + klo);
                t.u[1] = *(const uint4*)(xr + kt * 32 + klo + 8);
                A[kt] = t.v;
            }
        }

        // ---- Q0/K0/V0 = X @ W{q,k,v} via WMMA (B from LDS) ---------------
        for (int mat = 0; mat < 3; ++mat) {
            const unsigned short* Bm = &sW[mat * WMAT_ELEMS];
            __bf16* sp = (__bf16*)((mat == 0) ? sQ : (mat == 1) ? sK : sV);
            for (int nt = 0; nt < NTILE_N; ++nt) {
                v8f acc = {0.f, 0.f, 0.f, 0.f, 0.f, 0.f, 0.f, 0.f};
#pragma unroll
                for (int kt = 0; kt < NTILE_K; ++kt) {
                    union { uint4 u[2]; v16bf v; } tb;
                    const unsigned short* bp = Bm + ((kt * NTILE_N + nt) * 32 + lane) * 16;
                    tb.u[0] = *(const uint4*)(bp);
                    tb.u[1] = *(const uint4*)(bp + 8);
                    acc = __builtin_amdgcn_wmma_f32_16x16x32_bf16(
                              false, A[kt], false, tb.v, (short)0, acc, false, false);
                }
                const int c  = nt * 16 + (lane & 15);
                const int rb = (lane >> 4) * 8;
#pragma unroll
                for (int j = 0; j < 8; ++j) sp[(rb + j) * DP + c] = (__bf16)acc[j];
            }
        }
        asm volatile("s_wait_dscnt 0x0" ::: "memory");

        // ---- attention: 16 points x 8 heads = 128 (p,h) pairs, 4 passes --
#pragma unroll
        for (int it = 0; it < 4; ++it) {
            const int pair = it * 32 + lane;
            const int p = pair >> 3;
            const int h = pair & 7;
            int grow = row0 + p; if (grow >= m) grow = m - 1;
            const int t = tmod[grow];
            const float* q1 = TQ + (size_t)t * D_IN + h * DH;
            const float* k1 = TK + (size_t)t * D_IN + h * DH;
            const float* v1 = TV + (size_t)t * D_IN + h * DH;
            const __bf16* q0 = (const __bf16*)&sQ[p * DP + h * DH];
            const __bf16* k0 = (const __bf16*)&sK[p * DP + h * DH];
            const __bf16* v0 = (const __bf16*)&sV[p * DP + h * DH];

            float q0f[DH], k0f[DH], v0f[DH], q1f[DH], k1f[DH], v1f[DH];
#pragma unroll
            for (int d = 0; d < DH; ++d) {
                q0f[d] = (float)q0[d] + bq[h * DH + d];
                k0f[d] = (float)k0[d] + bk[h * DH + d];
                v0f[d] = (float)v0[d] + bv[h * DH + d];
                q1f[d] = q1[d]; k1f[d] = k1[d]; v1f[d] = v1[d];
            }
            float s00 = 0.f, s01 = 0.f, s10 = 0.f, s11 = 0.f;
#pragma unroll
            for (int d = 0; d < DH; ++d) {
                s00 += q0f[d] * k0f[d];  s01 += q0f[d] * k1f[d];
                s10 += q1f[d] * k0f[d];  s11 += q1f[d] * k1f[d];
            }
            const float sc = 0.25819888974716112f;   // 1/sqrt(15)
            s00 *= sc; s01 *= sc; s10 *= sc; s11 *= sc;
            float m0 = fmaxf(s00, s01);
            float e00 = __expf(s00 - m0), e01 = __expf(s01 - m0);
            float r0 = 1.0f / (e00 + e01);
            float m1 = fmaxf(s10, s11);
            float e10 = __expf(s10 - m1), e11 = __expf(s11 - m1);
            float r1 = 1.0f / (e10 + e11);
            const float c0 = 0.5f * (e00 * r0 + e10 * r1);   // weight on v0
            const float c1 = 0.5f * (e01 * r0 + e11 * r1);   // weight on v1
            __bf16* z = (__bf16*)&sZ[p * DP + h * DH];
#pragma unroll
            for (int d = 0; d < DH; ++d) z[d] = (__bf16)(c0 * v0f[d] + c1 * v1f[d]);
        }
        if (lane < 16) {               // zero Z pad columns 120..127
            uint4 zz; zz.x = zz.y = zz.z = zz.w = 0u;
            *(uint4*)(&sZ[lane * DP + D_IN]) = zz;
        }
        asm volatile("s_wait_dscnt 0x0" ::: "memory");

        // ---- out = Z @ Wo + bo (WMMA, A and B both from LDS) -------------
        v16bf AZ[NTILE_K];
        {
            const unsigned short* zr = &sZ[(lane & 15) * DP];
#pragma unroll
            for (int kt = 0; kt < NTILE_K; ++kt) {
                union { uint4 u[2]; v16bf v; } t;
                t.u[0] = *(const uint4*)(zr + kt * 32 + klo);
                t.u[1] = *(const uint4*)(zr + kt * 32 + klo + 8);
                AZ[kt] = t.v;
            }
        }
        const unsigned short* Bo = &sW[3 * WMAT_ELEMS];
        for (int nt = 0; nt < NTILE_N; ++nt) {
            v8f acc = {0.f, 0.f, 0.f, 0.f, 0.f, 0.f, 0.f, 0.f};
#pragma unroll
            for (int kt = 0; kt < NTILE_K; ++kt) {
                union { uint4 u[2]; v16bf v; } tb;
                const unsigned short* bp = Bo + ((kt * NTILE_N + nt) * 32 + lane) * 16;
                tb.u[0] = *(const uint4*)(bp);
                tb.u[1] = *(const uint4*)(bp + 8);
                acc = __builtin_amdgcn_wmma_f32_16x16x32_bf16(
                          false, AZ[kt], false, tb.v, (short)0, acc, false, false);
            }
            const int c = nt * 16 + (lane & 15);
            if (c < D_IN) {
                const float bias = bo[c];
                const int rb = (lane >> 4) * 8;
#pragma unroll
                for (int j = 0; j < 8; ++j) {
                    const int r = row0 + rb + j;
                    if (r < m) out[(size_t)r * D_IN + c] = acc[j] + bias;
                }
            }
        }
    }
}

// ---------------------------------------------------------------------------
extern "C" void kernel_launch(void* const* d_in, const int* in_sizes, int n_in,
                              void* d_out, int out_size, void* d_ws, size_t ws_size,
                              hipStream_t stream) {
    const float* q_pts  = (const float*)d_in[0];
    const int*   q_tms  = (const int*)d_in[1];
    const float* dyn    = (const float*)d_in[2];
    const float* temb   = (const float*)d_in[3];
    const int*   buf    = (const int*)d_in[4];
    const float* Wq = (const float*)d_in[5];  const float* bq = (const float*)d_in[6];
    const float* Wk = (const float*)d_in[7];  const float* bk = (const float*)d_in[8];
    const float* Wv = (const float*)d_in[9];  const float* bv = (const float*)d_in[10];
    const float* Wo = (const float*)d_in[11]; const float* bo = (const float*)d_in[12];
    float* out = (float*)d_out;
    const int m = in_sizes[0] / 3;

    // workspace layout
    char* ws = (char*)d_ws;
    auto align256 = [](size_t x) { return (x + 255) & ~(size_t)255; };
    const size_t PACK_BYTES = (size_t)4 * WMAT_ELEMS * 2;                    // 128 KB
    const size_t T_BYTES    = align256((size_t)MOD_T * D_IN * sizeof(float));
    size_t off = 0;
    unsigned short* packed = (unsigned short*)(ws + off); off = align256(off + PACK_BYTES);
    float* TQ   = (float*)(ws + off); off += T_BYTES;
    float* TK   = (float*)(ws + off); off += T_BYTES;
    float* TV   = (float*)(ws + off); off += T_BYTES;
    int*   tmod = (int*)(ws + off);   off = align256(off + (size_t)m * sizeof(int));
    unsigned short* X = (unsigned short*)(ws + off);       // m * 128 bf16

    // 1) pack weights into WMMA B-operand layout
    k_pack<<<(4 * NTILE_K * NTILE_N * 32 + 255) / 256, 256, 0, stream>>>(Wq, Wk, Wv, Wo, packed);
    // 2) precompute time-token projections (bias folded in)
    k_timeproj<<<(3 * MOD_T * D_IN + 255) / 256, 256, 0, stream>>>(
        temb, Wq, bq, Wk, bk, Wv, bv, TQ, TK, TV);
    // 3) hash gather + trilinear blend -> bf16 X rows + t_mod
    {
        int waves_per_block = 256 / 32;
        int blocks = (m + waves_per_block - 1) / waves_per_block;
        k_gather<<<blocks, 256, 0, stream>>>(q_pts, q_tms, dyn, buf, X, tmod, m);
    }
    // 4) fused WMMA projections + 2-token attention + output projection
    k_main<<<2048, 32, 0, stream>>>(X, tmod, packed, TQ, TK, TV, bq, bk, bv, bo, out, m);
}
